// AttentionSubsample_56865366999089
// MI455X (gfx1250) — matrix operand order repaired and checked by reading, not compiled, not run on hardware
//
#include <hip/hip_runtime.h>
#include <hip/hip_fp16.h>

typedef __attribute__((ext_vector_type(16))) _Float16 v16h;
typedef __attribute__((ext_vector_type(8)))  float    v8f;
typedef __attribute__((ext_vector_type(4)))  unsigned int v4u;
typedef __attribute__((ext_vector_type(8)))  unsigned int v8u;

#define BATCH   512
#define NPTS    196
#define NSUB    49
#define INDIM   256
#define OUTDIM  512
#define KEYD    16
#define NHEADS  8
#define DVAL    32
#define DHDIM   256
#define NHKD    128
#define KVOUT   384
#define ATTSCALE 0.25f     // 16^-0.5
#define NKPAD   224        // 196 padded to 7*32
#define MQPAD   64         // 49 padded to 64

#define WMMA16(a, b, c) __builtin_amdgcn_wmma_f32_16x16x32_f16(false, (a), false, (b), (short)0, (c), false, false)

// ---------------- Tensor Data Mover: contiguous 1-D f16 copy global -> LDS ----------------
// D# per CDNA5 ISA: group0 = {flags, lds_addr, global_addr, type=2},
// group1 = {data_size=2B, tensor_dim0=n, tensor_dim1=1, tile_dim0=n, tile_dim1=1, stride0=n}.
__device__ __forceinline__ void tdm_load_1d_f16(unsigned lds_off, const void* gptr, unsigned n) {
  unsigned long long ga = (unsigned long long)(size_t)gptr;
  v4u g0;
  g0[0] = 1u;                                        // count=1 (valid user descriptor)
  g0[1] = lds_off;                                   // lds_addr (bytes)
  g0[2] = (unsigned)(ga & 0xffffffffu);              // global_addr[31:0]
  g0[3] = (unsigned)((ga >> 32) & 0x01ffffffu)       // global_addr[56:32]
          | (2u << 30);                              // type=2 ("image")
  v8u g1;
  g1[0] = (1u << 16);                                // workgroup_mask=0, data_size=1 (2 bytes)
  g1[1] = (n & 0xffffu) << 16;                       // tensor_dim0[15:0]
  g1[2] = ((n >> 16) & 0xffffu) | (1u << 16);        // tensor_dim0[31:16] | tensor_dim1=1
  g1[3] = (n & 0xffffu) << 16;                       // tile_dim0 = n
  g1[4] = 1u;                                        // tile_dim1 = 1
  g1[5] = n;                                         // tensor_dim0_stride[31:0]
  g1[6] = 0u;
  g1[7] = 0u;
  asm volatile("tensor_load_to_lds %0, %1" :: "s"(g0), "s"(g1) : "memory");
}

// ---------------- small prep kernels ----------------

__global__ void cvt_f16_kernel(const float* __restrict__ in, _Float16* __restrict__ out, long long n) {
  long long i = (long long)blockIdx.x * blockDim.x + threadIdx.x;
  long long stride = (long long)gridDim.x * blockDim.x;
  for (; i < n; i += stride) out[i] = (_Float16)in[i];
}

__global__ void bn_prep_kernel(const float* __restrict__ g, const float* __restrict__ b,
                               const float* __restrict__ m, const float* __restrict__ v,
                               float* __restrict__ s, float* __restrict__ t, int n) {
  int i = blockIdx.x * blockDim.x + threadIdx.x;
  if (i < n) {
    float sc = g[i] * rsqrtf(v[i] + 1e-5f);
    s[i] = sc;
    t[i] = b[i] - m[i] * sc;
  }
}

__global__ void bias_mat_kernel(const float* __restrict__ tab, const int* __restrict__ idxs,
                                float* __restrict__ ab, int n_off) {
  int i = blockIdx.x * blockDim.x + threadIdx.x;
  const int total = NHEADS * NSUB * NPTS;
  if (i < total) {
    int h = i / (NSUB * NPTS);
    int r = i % (NSUB * NPTS);
    ab[i] = tab[h * n_off + idxs[r]];
  }
}

// ---------------- GEMM 1: kv = BN(x @ w_kv^T) -> k/v f16, per-head layout ----------------
// M = 100352 (6272 tiles), N = 384 (12 tile-pairs), K = 256; wave computes 16x32 (A reused)
__global__ void __launch_bounds__(256) kv_gemm_kernel(
    const _Float16* __restrict__ xh, const _Float16* __restrict__ wh,
    const float* __restrict__ bns, const float* __restrict__ bnt,
    _Float16* __restrict__ kf, _Float16* __restrict__ vf) {
  const int lane = threadIdx.x & 31;
  const int wave = threadIdx.x >> 5;
  const int NP = KVOUT / 32;              // 12 pairs
  int tile = blockIdx.x * 8 + wave;
  int mt = tile / NP, np = tile % NP;
  const int m15 = lane & 15;
  const int akb = (lane < 16) ? 0 : 16;
  const int bkb = (lane < 16) ? 0 : 16;
  const _Float16* arow  = xh + (size_t)(mt * 16 + m15) * INDIM;
  const _Float16* brow0 = wh + (size_t)(np * 32 + m15) * INDIM;
  const _Float16* brow1 = wh + (size_t)(np * 32 + 16 + m15) * INDIM;
  v8f c0 = {}, c1 = {};
#pragma unroll
  for (int k0 = 0; k0 < INDIM; k0 += 32) {
    v16h a, b0, b1;
    const _Float16* ap  = arow  + k0 + akb;
    const _Float16* bp0 = brow0 + k0 + bkb;
    const _Float16* bp1 = brow1 + k0 + bkb;
#pragma unroll
    for (int i = 0; i < 8; ++i) { a[i] = ap[i]; a[8 + i] = ap[8 + i]; }
#pragma unroll
    for (int i = 0; i < 16; ++i) { b0[i] = bp0[i]; b1[i] = bp1[i]; }
    c0 = WMMA16(a, b0, c0);
    c1 = WMMA16(a, b1, c1);
  }
#pragma unroll
  for (int t = 0; t < 2; ++t) {
    v8f cc = t ? c1 : c0;
    int ocol = np * 32 + t * 16 + m15;
    float s = bns[ocol], tt = bnt[ocol];
    int hh = ocol / (KEYD + DVAL);
    int j  = ocol % (KEYD + DVAL);
#pragma unroll
    for (int i = 0; i < 8; ++i) {
      int r = mt * 16 + ((lane < 16) ? i : 8 + i);
      int bidx = r / NPTS, nn = r % NPTS;
      float val = cc[i] * s + tt;
      if (j < KEYD)
        kf[(((size_t)bidx * NHEADS + hh) * NPTS + nn) * KEYD + j] = (_Float16)val;
      else
        vf[(((size_t)bidx * NHEADS + hh) * NPTS + nn) * DVAL + (j - KEYD)] = (_Float16)val;
    }
  }
}

// ---------------- GEMM 2: q = BN(xs @ w_q^T), strided subsample gather ----------------
// M = 25088 (1568 tiles), N = 128 (4 tile-pairs), K = 256
__global__ void __launch_bounds__(256) q_gemm_kernel(
    const _Float16* __restrict__ xh, const _Float16* __restrict__ wh,
    const float* __restrict__ bns, const float* __restrict__ bnt,
    _Float16* __restrict__ qf) {
  const int lane = threadIdx.x & 31;
  const int wave = threadIdx.x >> 5;
  const int NP = NHKD / 32;               // 4 pairs
  int tile = blockIdx.x * 8 + wave;
  int mt = tile / NP, np = tile % NP;
  const int m15 = lane & 15;
  int rflat = mt * 16 + m15;
  int bidx = rflat / NSUB, n_ = rflat % NSUB;
  int xrow = bidx * NPTS + (n_ / 7) * 2 * 14 + (n_ % 7) * 2;   // ::2,::2 subsample
  const int akb = (lane < 16) ? 0 : 16;
  const int bkb = (lane < 16) ? 0 : 16;
  const _Float16* arow  = xh + (size_t)xrow * INDIM;
  const _Float16* brow0 = wh + (size_t)(np * 32 + m15) * INDIM;
  const _Float16* brow1 = wh + (size_t)(np * 32 + 16 + m15) * INDIM;
  v8f c0 = {}, c1 = {};
#pragma unroll
  for (int k0 = 0; k0 < INDIM; k0 += 32) {
    v16h a, b0, b1;
    const _Float16* ap  = arow  + k0 + akb;
    const _Float16* bp0 = brow0 + k0 + bkb;
    const _Float16* bp1 = brow1 + k0 + bkb;
#pragma unroll
    for (int i = 0; i < 8; ++i) { a[i] = ap[i]; a[8 + i] = ap[8 + i]; }
#pragma unroll
    for (int i = 0; i < 16; ++i) { b0[i] = bp0[i]; b1[i] = bp1[i]; }
    c0 = WMMA16(a, b0, c0);
    c1 = WMMA16(a, b1, c1);
  }
#pragma unroll
  for (int t = 0; t < 2; ++t) {
    v8f cc = t ? c1 : c0;
    int ocol = np * 32 + t * 16 + m15;
    float s = bns[ocol], tt = bnt[ocol];
    int hh = ocol / KEYD, j = ocol % KEYD;
#pragma unroll
    for (int i = 0; i < 8; ++i) {
      int r = mt * 16 + ((lane < 16) ? i : 8 + i);
      int rb = r / NSUB, rn = r % NSUB;
      float val = cc[i] * s + tt;
      qf[(((size_t)rb * NHEADS + hh) * NSUB + rn) * KEYD + j] = (_Float16)val;
    }
  }
}

// ---------------- Fused attention per (b,h) ----------------
__global__ void __launch_bounds__(256) attn_kernel(
    const _Float16* __restrict__ qf, const _Float16* __restrict__ kf,
    const _Float16* __restrict__ vf, const float* __restrict__ ab,
    _Float16* __restrict__ oh) {
  __shared__ _Float16 sQ[MQPAD * 32];        //  4 KB, keydim zero-padded to 32
  __shared__ _Float16 sK[NKPAD * KEYD];      //  7 KB
  __shared__ _Float16 sV[NKPAD * DVAL];      // 14 KB
  __shared__ _Float16 sP[MQPAD * NKPAD];     // 28 KB (logits -> probs)
  const int b = blockIdx.x / NHEADS;
  const int h = blockIdx.x % NHEADS;
  const int tid = threadIdx.x;
  const int lane = tid & 31, wave = tid >> 5;
  const int m15 = lane & 15;
  const bool blo = (lane < 16);
  const int akb = blo ? 0 : 16;

  const _Float16* qp = qf + ((size_t)b * NHEADS + h) * NSUB * KEYD;
  const _Float16* kp = kf + ((size_t)b * NHEADS + h) * NPTS * KEYD;
  const _Float16* vp = vf + ((size_t)b * NHEADS + h) * NPTS * DVAL;

  // TDM async copy of k (196x16) and v (196x32) into LDS, issued by one wave.
  if (tid == 0) {
    tdm_load_1d_f16((unsigned)(size_t)(void*)sK, kp, NPTS * KEYD);
    tdm_load_1d_f16((unsigned)(size_t)(void*)sV, vp, NPTS * DVAL);
    __builtin_amdgcn_s_wait_tensorcnt(0);
  }
  // q with interleaved keydim zero-pad, plus zeroing of key-pad tails.
  for (int i = tid; i < MQPAD * 32; i += 256) {
    int r = i >> 5, kk = i & 31;
    sQ[i] = (r < NSUB && kk < KEYD) ? qp[r * KEYD + kk] : (_Float16)0.f;
  }
  for (int i = NPTS * KEYD + tid; i < NKPAD * KEYD; i += 256) sK[i] = (_Float16)0.f;
  for (int i = NPTS * DVAL + tid; i < NKPAD * DVAL; i += 256) sV[i] = (_Float16)0.f;
  __syncthreads();

  // ---- S = q @ k^T : 4 Mtiles x 14 Ntiles = 56 tiles over 8 waves ----
  for (int t = wave; t < 4 * 14; t += 8) {
    int mt = t / 14, nt = t % 14;
    v16h a, bm;
#pragma unroll
    for (int i = 0; i < 8; ++i) {
      a[i]     = sQ[(mt * 16 + m15) * 32 + akb + i];
      a[8 + i] = sQ[(mt * 16 + m15) * 32 + akb + 8 + i];
    }
    if (blo) {                                // K=0..15 live, 16..31 zero pad
#pragma unroll
      for (int i = 0; i < 16; ++i) bm[i] = sK[(nt * 16 + m15) * KEYD + i];
    } else {
#pragma unroll
      for (int i = 0; i < 16; ++i) bm[i] = (_Float16)0.f;
    }
    v8f c = {};
    c = WMMA16(a, bm, c);
    int kk = nt * 16 + m15;
#pragma unroll
    for (int i = 0; i < 8; ++i) {
      int r = mt * 16 + (blo ? i : 8 + i);
      float logit;
      if (kk < NPTS)
        logit = c[i] * ATTSCALE + ((r < NSUB) ? ab[(h * NSUB + r) * NPTS + kk] : 0.f);
      else
        logit = -1e30f;                       // padded keys -> prob 0
      sP[r * NKPAD + kk] = (_Float16)logit;
    }
  }
  __syncthreads();

  // ---- softmax over keys: 4 threads per row, wave32 shuffle reduce ----
  {
    int row = tid >> 2, sub = tid & 3;
    float mx = -1e30f;
    for (int kk = sub; kk < NKPAD; kk += 4) mx = fmaxf(mx, (float)sP[row * NKPAD + kk]);
    mx = fmaxf(mx, __shfl_xor(mx, 1, 32));
    mx = fmaxf(mx, __shfl_xor(mx, 2, 32));
    float sum = 0.f;
    for (int kk = sub; kk < NKPAD; kk += 4) {
      float e = __expf((float)sP[row * NKPAD + kk] - mx);
      sP[row * NKPAD + kk] = (_Float16)e;
      sum += e;
    }
    sum += __shfl_xor(sum, 1, 32);
    sum += __shfl_xor(sum, 2, 32);
    float inv = 1.f / sum;
    for (int kk = sub; kk < NKPAD; kk += 4)
      sP[row * NKPAD + kk] = (_Float16)((float)sP[row * NKPAD + kk] * inv);
  }
  __syncthreads();

  // ---- O = P @ V : 8 tiles, one per wave; K loop 224/32 ----
  {
    int mt = wave >> 1, nt = wave & 1;
    const int bkb = blo ? 0 : 16;
    v8f c = {};
    for (int kc = 0; kc < NKPAD / 32; ++kc) {
      v16h a, bm;
      int base = kc * 32;
#pragma unroll
      for (int i = 0; i < 8; ++i) {
        a[i]     = sP[(mt * 16 + m15) * NKPAD + base + akb + i];
        a[8 + i] = sP[(mt * 16 + m15) * NKPAD + base + akb + 8 + i];
      }
#pragma unroll
      for (int i = 0; i < 16; ++i) bm[i] = sV[(base + bkb + i) * DVAL + nt * 16 + m15];
      c = WMMA16(a, bm, c);
    }
    int d = nt * 16 + m15;
#pragma unroll
    for (int i = 0; i < 8; ++i) {
      int r = mt * 16 + (blo ? i : 8 + i);
      if (r < NSUB) {
        float val = c[i];
        float hs = val * fminf(fmaxf(val + 3.f, 0.f), 6.f) * (1.f / 6.f);  // hardswish
        oh[((size_t)b * NSUB + r) * DHDIM + h * DVAL + d] = (_Float16)hs;
      }
    }
  }
}

// ---------------- GEMM 3: out = BN(hardswish(O) @ w_p^T), f32 output ----------------
// M = 25088 (1568 tiles), N = 512 (16 tile-pairs), K = 256
__global__ void __launch_bounds__(256) proj_gemm_kernel(
    const _Float16* __restrict__ oh, const _Float16* __restrict__ wh,
    const float* __restrict__ bns, const float* __restrict__ bnt,
    float* __restrict__ out) {
  const int lane = threadIdx.x & 31;
  const int wave = threadIdx.x >> 5;
  const int NP = OUTDIM / 32;             // 16 pairs
  int tile = blockIdx.x * 8 + wave;
  int mt = tile / NP, np = tile % NP;
  const int m15 = lane & 15;
  const int akb = (lane < 16) ? 0 : 16;
  const int bkb = (lane < 16) ? 0 : 16;
  const _Float16* arow  = oh + (size_t)(mt * 16 + m15) * DHDIM;
  const _Float16* brow0 = wh + (size_t)(np * 32 + m15) * DHDIM;
  const _Float16* brow1 = wh + (size_t)(np * 32 + 16 + m15) * DHDIM;
  v8f c0 = {}, c1 = {};
#pragma unroll
  for (int k0 = 0; k0 < DHDIM; k0 += 32) {
    v16h a, b0, b1;
    const _Float16* ap  = arow  + k0 + akb;
    const _Float16* bp0 = brow0 + k0 + bkb;
    const _Float16* bp1 = brow1 + k0 + bkb;
#pragma unroll
    for (int i = 0; i < 8; ++i) { a[i] = ap[i]; a[8 + i] = ap[8 + i]; }
#pragma unroll
    for (int i = 0; i < 16; ++i) { b0[i] = bp0[i]; b1[i] = bp1[i]; }
    c0 = WMMA16(a, b0, c0);
    c1 = WMMA16(a, b1, c1);
  }
#pragma unroll
  for (int t = 0; t < 2; ++t) {
    v8f cc = t ? c1 : c0;
    int ocol = np * 32 + t * 16 + m15;
    float s = bns[ocol], tt = bnt[ocol];
#pragma unroll
    for (int i = 0; i < 8; ++i) {
      int r = mt * 16 + ((lane < 16) ? i : 8 + i);
      out[(size_t)r * OUTDIM + ocol] = cc[i] * s + tt;
    }
  }
}

// ---------------- launch ----------------

extern "C" void kernel_launch(void* const* d_in, const int* in_sizes, int n_in,
                              void* d_out, int out_size, void* d_ws, size_t ws_size,
                              hipStream_t stream) {
  const float* x     = (const float*)d_in[0];
  const float* w_kv  = (const float*)d_in[1];
  const float* kv_g  = (const float*)d_in[2];
  const float* kv_b  = (const float*)d_in[3];
  const float* kv_m  = (const float*)d_in[4];
  const float* kv_v  = (const float*)d_in[5];
  const float* w_q   = (const float*)d_in[6];
  const float* q_g   = (const float*)d_in[7];
  const float* q_b   = (const float*)d_in[8];
  const float* q_m   = (const float*)d_in[9];
  const float* q_v   = (const float*)d_in[10];
  const float* w_p   = (const float*)d_in[11];
  const float* p_g   = (const float*)d_in[12];
  const float* p_b   = (const float*)d_in[13];
  const float* p_m   = (const float*)d_in[14];
  const float* p_v   = (const float*)d_in[15];
  const float* ab_t  = (const float*)d_in[16];
  const int*   bidx  = (const int*)d_in[17];
  float* out = (float*)d_out;

  char* ws = (char*)d_ws;
  size_t off = 0;
  auto walloc = [&](size_t bytes) -> void* {
    void* p = ws + off;
    off = (off + bytes + 255) & ~(size_t)255;
    return p;
  };
  _Float16* xh   = (_Float16*)walloc((size_t)BATCH * NPTS * INDIM * 2);
  _Float16* wkvh = (_Float16*)walloc((size_t)KVOUT * INDIM * 2);
  _Float16* wqh  = (_Float16*)walloc((size_t)NHKD * INDIM * 2);
  _Float16* wph  = (_Float16*)walloc((size_t)OUTDIM * DHDIM * 2);
  _Float16* kf   = (_Float16*)walloc((size_t)BATCH * NHEADS * NPTS * KEYD * 2);
  _Float16* vf   = (_Float16*)walloc((size_t)BATCH * NHEADS * NPTS * DVAL * 2);
  _Float16* qf   = (_Float16*)walloc((size_t)BATCH * NHEADS * NSUB * KEYD * 2);
  _Float16* ohb  = (_Float16*)walloc((size_t)BATCH * NSUB * DHDIM * 2);
  float*    ab   = (float*)walloc((size_t)NHEADS * NSUB * NPTS * 4);
  float* bnkv_s  = (float*)walloc(KVOUT * 4);
  float* bnkv_t  = (float*)walloc(KVOUT * 4);
  float* bnq_s   = (float*)walloc(NHKD * 4);
  float* bnq_t   = (float*)walloc(NHKD * 4);
  float* bnp_s   = (float*)walloc(OUTDIM * 4);
  float* bnp_t   = (float*)walloc(OUTDIM * 4);

  // one-time f32 -> f16 conversions (removes inner-loop cvt + halves GEMM read traffic)
  cvt_f16_kernel<<<4096, 256, 0, stream>>>(x, xh, (long long)BATCH * NPTS * INDIM);
  cvt_f16_kernel<<<96, 256, 0, stream>>>(w_kv, wkvh, (long long)KVOUT * INDIM);
  cvt_f16_kernel<<<32, 256, 0, stream>>>(w_q, wqh, (long long)NHKD * INDIM);
  cvt_f16_kernel<<<128, 256, 0, stream>>>(w_p, wph, (long long)OUTDIM * DHDIM);

  bn_prep_kernel<<<(KVOUT + 255) / 256, 256, 0, stream>>>(kv_g, kv_b, kv_m, kv_v, bnkv_s, bnkv_t, KVOUT);
  bn_prep_kernel<<<(NHKD + 255) / 256, 256, 0, stream>>>(q_g, q_b, q_m, q_v, bnq_s, bnq_t, NHKD);
  bn_prep_kernel<<<(OUTDIM + 255) / 256, 256, 0, stream>>>(p_g, p_b, p_m, p_v, bnp_s, bnp_t, OUTDIM);
  int n_off = in_sizes[16] / NHEADS;
  bias_mat_kernel<<<(NHEADS * NSUB * NPTS + 255) / 256, 256, 0, stream>>>(ab_t, bidx, ab, n_off);

  // kv GEMM: 6272 Mtiles * 12 pairs / 8 waves
  kv_gemm_kernel<<<(BATCH * NPTS / 16) * (KVOUT / 32) / 8, 256, 0, stream>>>(xh, wkvh, bnkv_s, bnkv_t, kf, vf);
  // q GEMM: 1568 * 4 / 8
  q_gemm_kernel<<<(BATCH * NSUB / 16) * (NHKD / 32) / 8, 256, 0, stream>>>(xh, wqh, bnq_s, bnq_t, qf);
  // attention: one block per (b,h)
  attn_kernel<<<BATCH * NHEADS, 256, 0, stream>>>(qf, kf, vf, ab, ohb);
  // projection GEMM: 1568 * 16 / 8
  proj_gemm_kernel<<<(BATCH * NSUB / 16) * (OUTDIM / 32) / 8, 256, 0, stream>>>(ohb, wph, bnp_s, bnp_t, out);
}
